// GCN_10548439679260
// MI455X (gfx1250) — matrix operand order repaired and checked by reading, not compiled, not run on hardware
//
#include <hip/hip_runtime.h>
#include <hip/hip_bf16.h>
#include <math.h>

// ---------------------------------------------------------------------------
// Model constants (match reference)
// ---------------------------------------------------------------------------
#define NUM_ROI   400
#define N_GRAPHS  128
#define EMB       128
#define RO_OUT    8
#define FC1_IN    (NUM_ROI * RO_OUT)   // 3200
#define K1_PAD    416                  // 400 padded to multiple of 32
#define N_FC1_PAD 448                  // 400 padded to multiple of 64
#define BN_EPS    1e-5f

typedef __attribute__((ext_vector_type(16))) __bf16 v16bf;
typedef __attribute__((ext_vector_type(8)))  __bf16 v8bf;
typedef __attribute__((ext_vector_type(8)))  float  v8f;

__device__ __forceinline__ v16bf cat8(v8bf lo, v8bf hi) {
    return __builtin_shufflevector(lo, hi, 0, 1, 2, 3, 4, 5, 6, 7,
                                           8, 9, 10, 11, 12, 13, 14, 15);
}

__device__ __forceinline__ float mishf(float x) {
    float sp = (x > 20.0f) ? x : log1pf(__expf(x));
    return x * tanhf(sp);
}

// Fragment loaders (layouts per ISA 7.12.2, wave32):
//   A 16x32: lane L row M=L&15; elems 0-7 -> K=k0+(L>>4)*8+i, elems 8-15 -> +16
//   B 32x16: lane L col N=L&15; elems 0-7 -> K=k0+(L>>4)*8+i, elems 8-15 -> +16
__device__ __forceinline__ v16bf load_afrag(const __bf16* Ar, int k0, int hi) {
    const int ka = k0 + (hi << 3);
    return cat8(*(const v8bf*)(Ar + ka), *(const v8bf*)(Ar + ka + 16));
}
__device__ __forceinline__ v16bf load_bfrag(const __bf16* Br, int k0, int hi) {
    const int kb = k0 + (hi << 3);
    return cat8(*(const v8bf*)(Br + kb), *(const v8bf*)(Br + kb + 16));
}

// ---------------------------------------------------------------------------
// A-matrix conversion: row-major f32 [rows,K] -> row-major bf16 [rows,Kpad]
// ---------------------------------------------------------------------------
__global__ void k_convert_padA(const float* __restrict__ src,
                               __bf16* __restrict__ dst,
                               int rows, int K, int Kpad) {
    long idx = (long)blockIdx.x * blockDim.x + threadIdx.x;
    long total = (long)rows * Kpad;
    if (idx >= total) return;
    int r = (int)(idx / Kpad);
    int k = (int)(idx % Kpad);
    dst[idx] = (k < K) ? (__bf16)src[(size_t)r * K + k] : (__bf16)0.0f;
}

// ---------------------------------------------------------------------------
// B-matrix conversion + transpose: f32 [K,N] -> bf16 Bt [Npad,Kpad]
// ---------------------------------------------------------------------------
__global__ void k_convert_transB(const float* __restrict__ src,
                                 __bf16* __restrict__ dst,
                                 int K, int N, int Kpad, int Npad) {
    long idx = (long)blockIdx.x * blockDim.x + threadIdx.x;
    long total = (long)Npad * Kpad;
    if (idx >= total) return;
    int n = (int)(idx / Kpad);
    int k = (int)(idx % Kpad);
    dst[idx] = (n < N && k < K) ? (__bf16)src[(size_t)k * N + n] : (__bf16)0.0f;
}

// ---------------------------------------------------------------------------
// Degree / normalization
// ---------------------------------------------------------------------------
__global__ void k_deg_init(float* __restrict__ deg, int n) {
    int i = blockIdx.x * blockDim.x + threadIdx.x;
    if (i < n) deg[i] = 1.0f;    // self-loop weight
}

__global__ void k_deg_accum(const int* __restrict__ col,
                            const float* __restrict__ ew,
                            float* __restrict__ deg, int E) {
    int e = blockIdx.x * blockDim.x + threadIdx.x;
    if (e < E) atomicAdd(&deg[col[e]], ew[e]);
}

__global__ void k_deg_finalize(float* __restrict__ deg, int n) {
    int i = blockIdx.x * blockDim.x + threadIdx.x;
    if (i < n) {
        float d = deg[i];
        deg[i] = (d > 0.0f) ? rsqrtf(d) : 0.0f;
    }
}

// ---------------------------------------------------------------------------
// bf16 WMMA GEMM:  C[M,NC] = A[M,KPAD] @ Bt[NC,KPAD]^T
// KPAD/NC compile-time: constant strides -> one B base register + immediate
// offsets; no spills under __launch_bounds__(128,1). One wave = 16x64 C slab.
// 2x-unrolled ping-pong pipeline: load S1(k+32) -> 4 WMMA(S0) ->
// load S0(k+64) -> 4 WMMA(S1). Roles alternate, so the loop-carried fragment
// sets need no register copies (no v_mov rotation), and each WMMA group runs
// against a partial s_wait_loadcnt with ~10 b128 loads still in flight.
// Tail parity (NSTEP = KPAD/32) handled with if constexpr; EXEC all-ones
// through every v_wmma (ISA 7.12).
// ---------------------------------------------------------------------------
template <int KPAD, int NC>
__global__ __launch_bounds__(128, 1)
void k_wmma_gemm_bf16(const __bf16* __restrict__ A,
                      const __bf16* __restrict__ Bt,
                      float* __restrict__ C, int M) {
    constexpr int GRPS  = NC >> 6;             // groups of 4 N-tiles
    constexpr int NSTEP = KPAD / 32;           // k-steps (13 / 4 / 100)
    const int lane  = threadIdx.x & 31;
    const int wave  = threadIdx.x >> 5;
    const int tile  = blockIdx.x * (blockDim.x >> 5) + wave;
    const int total = (M >> 4) * GRPS;
    if (tile >= total) return;                 // wave-uniform (grids divide exactly)

    const int tm   = tile / GRPS;
    const int g    = tile % GRPS;
    const int row0 = tm << 4;
    const int col0 = g << 6;
    const int hi   = lane >> 4;                // 0 or 1
    const int lo   = lane & 15;

    const __bf16* Ar = A + (size_t)(row0 + lo) * KPAD;
    const __bf16* Br = Bt + (size_t)(col0 + lo) * KPAD;   // single B base

    v8f acc0 = {}, acc1 = {}, acc2 = {}, acc3 = {};

    struct FragSet { v16bf a, b0, b1, b2, b3; };
    auto load_set = [&](int k0) {
        FragSet s;
        s.a  = load_afrag(Ar, k0, hi);
        s.b0 = load_bfrag(Br,             k0, hi);
        s.b1 = load_bfrag(Br + 16 * KPAD, k0, hi);
        s.b2 = load_bfrag(Br + 32 * KPAD, k0, hi);
        s.b3 = load_bfrag(Br + 48 * KPAD, k0, hi);
        return s;
    };
    auto mma = [&](const FragSet& s) {
        acc0 = __builtin_amdgcn_wmma_f32_16x16x32_bf16(false, s.a, false, s.b0, (short)0, acc0, false, false);
        acc1 = __builtin_amdgcn_wmma_f32_16x16x32_bf16(false, s.a, false, s.b1, (short)0, acc1, false, false);
        acc2 = __builtin_amdgcn_wmma_f32_16x16x32_bf16(false, s.a, false, s.b2, (short)0, acc2, false, false);
        acc3 = __builtin_amdgcn_wmma_f32_16x16x32_bf16(false, s.a, false, s.b3, (short)0, acc3, false, false);
    };

    FragSet s0 = load_set(0);
    int step = 0;
    #pragma unroll 1
    for (; step + 2 < NSTEP; step += 2) {
        if (step + 3 < NSTEP)
            __builtin_prefetch(Ar + (step + 3) * 32, 0, 3);  // global_prefetch_b8
        FragSet s1 = load_set((step + 1) * 32);
        mma(s0);                                 // overlaps S1 loads
        s0 = load_set((step + 2) * 32);
        mma(s1);                                 // overlaps new S0 loads
    }
    if constexpr ((NSTEP & 1) == 0) {            // two k-steps remain
        FragSet s1 = load_set((NSTEP - 1) * 32);
        mma(s0);
        mma(s1);
    } else {                                     // one k-step remains
        mma(s0);
    }

    float* Cr = C + (size_t)row0 * NC + col0 + lo;
    #pragma unroll
    for (int r = 0; r < 8; ++r) {
        float* crow = Cr + (size_t)(r + (hi << 3)) * NC;
        crow[0]  = acc0[r];
        crow[16] = acc1[r];
        crow[32] = acc2[r];
        crow[48] = acc3[r];
    }
}

// ---------------------------------------------------------------------------
// Self-loop init of aggregate: out[i,:] = dinv[i]^2 * h[i,:]   (F = EMB = 128)
// ---------------------------------------------------------------------------
__global__ void k_self_loop_init(const float* __restrict__ h,
                                 const float* __restrict__ dinv,
                                 float* __restrict__ out, long total) {
    long idx = (long)blockIdx.x * blockDim.x + threadIdx.x;
    if (idx < total) {
        int node = (int)(idx >> 7);            // /EMB
        float s = dinv[node];
        out[idx] = s * s * h[idx];
    }
}

// ---------------------------------------------------------------------------
// Edge scatter: one wave per edge, float4 per lane (EMB = 128).
// out[col,:] += dinv[row]*ew*dinv[col] * h[row,:]   (f32 atomics, L2-resident)
// ---------------------------------------------------------------------------
__global__ void k_edge_scatter(const int* __restrict__ row,
                               const int* __restrict__ col,
                               const float* __restrict__ ew,
                               const float* __restrict__ dinv,
                               const float* __restrict__ h,
                               float* __restrict__ out, int E) {
    int e    = blockIdx.x * (blockDim.x >> 5) + (threadIdx.x >> 5);
    int lane = threadIdx.x & 31;
    if (e >= E) return;
    int r = row[e];
    int c = col[e];
    float norm = dinv[r] * ew[e] * dinv[c];
    const float4 hv = ((const float4*)(h + (size_t)r * EMB))[lane];
    float* dst = out + (size_t)c * EMB + lane * 4;
    atomicAdd(dst + 0, norm * hv.x);
    atomicAdd(dst + 1, norm * hv.y);
    atomicAdd(dst + 2, norm * hv.z);
    atomicAdd(dst + 3, norm * hv.w);
}

// ---------------------------------------------------------------------------
// bias + mish -> bf16 (input to next GEMM), F = EMB = 128
// ---------------------------------------------------------------------------
__global__ void k_bias_mish_bf16(const float* __restrict__ in,
                                 const float* __restrict__ bias,
                                 __bf16* __restrict__ out, long total) {
    long i = (long)blockIdx.x * blockDim.x + threadIdx.x;
    if (i < total) {
        int j = (int)(i & (EMB - 1));
        out[i] = (__bf16)mishf(in[i] + bias[j]);
    }
}

// ---------------------------------------------------------------------------
// Readout: feats[n, 0:8] = mish(h[n,:] @ ro_w + ro_b) as bf16. Thread/node.
// ---------------------------------------------------------------------------
__global__ void k_readout(const __bf16* __restrict__ h,
                          const float* __restrict__ w,
                          const float* __restrict__ b,
                          __bf16* __restrict__ feats, int n) {
    int node = blockIdx.x * blockDim.x + threadIdx.x;
    if (node >= n) return;
    float acc[RO_OUT];
    #pragma unroll
    for (int k = 0; k < RO_OUT; ++k) acc[k] = b[k];
    const v8bf* hr = (const v8bf*)(h + (size_t)node * EMB);
    #pragma unroll 4
    for (int c = 0; c < EMB / 8; ++c) {
        v8bf hv = hr[c];
        #pragma unroll
        for (int u = 0; u < 8; ++u) {
            float hf = (float)hv[u];
            const float* wr = w + (c * 8 + u) * RO_OUT;
            #pragma unroll
            for (int k = 0; k < RO_OUT; ++k) acc[k] += hf * wr[k];
        }
    }
    #pragma unroll
    for (int k = 0; k < RO_OUT; ++k)
        feats[(size_t)node * RO_OUT + k] = (__bf16)mishf(acc[k]);
}

// ---------------------------------------------------------------------------
// BatchNorm (training stats over 128 graphs) + mish; fc1 bias folded in.
// One block per feature f (400 blocks, 128 threads = graphs). z has ldz cols.
// ---------------------------------------------------------------------------
__global__ void k_bn_mish(const float* __restrict__ z,
                          const float* __restrict__ fc1_b,
                          const float* __restrict__ gamma,
                          const float* __restrict__ beta,
                          float* __restrict__ zm, int ldz) {
    int f = blockIdx.x;
    int g = threadIdx.x;
    __shared__ float red[N_GRAPHS];

    float v = z[(size_t)g * ldz + f] + fc1_b[f];

    red[g] = v;
    __syncthreads();
    for (int s = N_GRAPHS / 2; s > 0; s >>= 1) {
        if (g < s) red[g] += red[g + s];
        __syncthreads();
    }
    float mu = red[0] * (1.0f / N_GRAPHS);
    __syncthreads();

    float d = v - mu;
    red[g] = d * d;
    __syncthreads();
    for (int s = N_GRAPHS / 2; s > 0; s >>= 1) {
        if (g < s) red[g] += red[g + s];
        __syncthreads();
    }
    float var = red[0] * (1.0f / N_GRAPHS);

    float zn = d * rsqrtf(var + BN_EPS) * gamma[f] + beta[f];
    zm[(size_t)g * NUM_ROI + f] = mishf(zn);
}

// ---------------------------------------------------------------------------
// fc2: logits[128,2] = zm[128,400] @ w[400,2] + b. Single 256-thread block.
// ---------------------------------------------------------------------------
__global__ void k_fc2(const float* __restrict__ zm,
                      const float* __restrict__ w,
                      const float* __restrict__ b,
                      float* __restrict__ out) {
    int t = threadIdx.x;
    int g = t >> 1;
    int c = t & 1;
    float acc = b[c];
    const float* zr = zm + (size_t)g * NUM_ROI;
    for (int j = 0; j < NUM_ROI; ++j) acc += zr[j] * w[j * 2 + c];
    out[g * 2 + c] = acc;
}

// ---------------------------------------------------------------------------
// Host-side launch
// ---------------------------------------------------------------------------
static inline size_t alup(size_t x) { return (x + 255) & ~(size_t)255; }

extern "C" void kernel_launch(void* const* d_in, const int* in_sizes, int n_in,
                              void* d_out, int out_size, void* d_ws, size_t ws_size,
                              hipStream_t stream) {
    const float* x     = (const float*)d_in[0];
    const int*   ei    = (const int*)  d_in[1];
    const float* ew    = (const float*)d_in[2];
    /* d_in[3] = batch: implied by contiguous reshape, unused */
    const float* w1    = (const float*)d_in[4];
    const float* b1    = (const float*)d_in[5];
    const float* w2    = (const float*)d_in[6];
    const float* b2    = (const float*)d_in[7];
    const float* ro_w  = (const float*)d_in[8];
    const float* ro_b  = (const float*)d_in[9];
    const float* fc1w  = (const float*)d_in[10];
    const float* fc1b  = (const float*)d_in[11];
    const float* gamma = (const float*)d_in[12];
    const float* beta  = (const float*)d_in[13];
    const float* fc2w  = (const float*)d_in[14];
    const float* fc2b  = (const float*)d_in[15];

    const int NN = in_sizes[0] / NUM_ROI;   // 51200 nodes
    const int E  = in_sizes[1] / 2;         // 2048000 edges
    const int* row = ei;
    const int* col = ei + E;

    // ---- workspace arena ----
    char* ws = (char*)d_ws;
    size_t o = 0;
    float*  dinv   = (float*) (ws + o); o += alup((size_t)NN * 4);
    __bf16* xb     = (__bf16*)(ws + o); o += alup((size_t)NN * K1_PAD * 2);
    float*  hbuf   = (float*) (ws + o); o += alup((size_t)NN * EMB * 4);
    float*  abuf   = (float*) (ws + o); o += alup((size_t)NN * EMB * 4);
    __bf16* hb     = (__bf16*)(ws + o); o += alup((size_t)NN * EMB * 2);
    __bf16* w1t    = (__bf16*)(ws + o); o += alup((size_t)EMB * K1_PAD * 2);
    __bf16* w2t    = (__bf16*)(ws + o); o += alup((size_t)EMB * EMB * 2);
    __bf16* fc1wt  = (__bf16*)(ws + o); o += alup((size_t)N_FC1_PAD * FC1_IN * 2);
    __bf16* featsb = (__bf16*)(ws + o); o += alup((size_t)N_GRAPHS * FC1_IN * 2);
    float*  zbuf   = (float*) (ws + o); o += alup((size_t)N_GRAPHS * N_FC1_PAD * 4);
    float*  zm     = (float*) (ws + o); o += alup((size_t)N_GRAPHS * NUM_ROI * 4);
    (void)ws_size; (void)n_in; (void)out_size;

    const long nodeF = (long)NN * EMB;

    // ---- conversions (pad K / transpose B) ----
    {
        long n = (long)NN * K1_PAD;
        k_convert_padA<<<(int)((n + 255) / 256), 256, 0, stream>>>(x, xb, NN, NUM_ROI, K1_PAD);
        n = (long)EMB * K1_PAD;
        k_convert_transB<<<(int)((n + 255) / 256), 256, 0, stream>>>(w1, w1t, NUM_ROI, EMB, K1_PAD, EMB);
        n = (long)EMB * EMB;
        k_convert_transB<<<(int)((n + 255) / 256), 256, 0, stream>>>(w2, w2t, EMB, EMB, EMB, EMB);
        n = (long)N_FC1_PAD * FC1_IN;
        k_convert_transB<<<(int)((n + 255) / 256), 256, 0, stream>>>(fc1w, fc1wt, FC1_IN, NUM_ROI, FC1_IN, N_FC1_PAD);
    }

    // ---- degree normalization ----
    k_deg_init    <<<(NN + 255) / 256, 256, 0, stream>>>(dinv, NN);
    k_deg_accum   <<<(E + 255) / 256, 256, 0, stream>>>(col, ew, dinv, E);
    k_deg_finalize<<<(NN + 255) / 256, 256, 0, stream>>>(dinv, NN);

    // ---- conv1: h = x @ W1  (M=51200, NC=128, KPAD=416) ----
    {
        int waves = (NN >> 4) * (EMB >> 6);             // 6400
        k_wmma_gemm_bf16<K1_PAD, EMB><<<waves / 4, 128, 0, stream>>>(xb, w1t, hbuf, NN);
    }
    k_self_loop_init<<<(int)((nodeF + 255) / 256), 256, 0, stream>>>(hbuf, dinv, abuf, nodeF);
    k_edge_scatter  <<<(E + 7) / 8, 256, 0, stream>>>(row, col, ew, dinv, hbuf, abuf, E);
    k_bias_mish_bf16<<<(int)((nodeF + 255) / 256), 256, 0, stream>>>(abuf, b1, hb, nodeF);

    // ---- conv2: h = act1 @ W2  (M=51200, NC=128, KPAD=128) ----
    {
        int waves = (NN >> 4) * (EMB >> 6);
        k_wmma_gemm_bf16<EMB, EMB><<<waves / 4, 128, 0, stream>>>(hb, w2t, hbuf, NN);
    }
    k_self_loop_init<<<(int)((nodeF + 255) / 256), 256, 0, stream>>>(hbuf, dinv, abuf, nodeF);
    k_edge_scatter  <<<(E + 7) / 8, 256, 0, stream>>>(row, col, ew, dinv, hbuf, abuf, E);
    k_bias_mish_bf16<<<(int)((nodeF + 255) / 256), 256, 0, stream>>>(abuf, b2, hb, nodeF);

    // ---- readout -> feats (bf16, [128, 3200] contiguous reshape) ----
    k_readout<<<(NN + 255) / 256, 256, 0, stream>>>(hb, ro_w, ro_b, featsb, NN);

    // ---- fc1: z = feats @ fc1_w  (M=128, NC=448 padded, KPAD=3200) ----
    {
        int waves = (N_GRAPHS >> 4) * (N_FC1_PAD >> 6);  // 8*7 = 56
        k_wmma_gemm_bf16<FC1_IN, N_FC1_PAD><<<waves / 4, 128, 0, stream>>>(featsb, fc1wt, zbuf, N_GRAPHS);
    }

    // ---- BN (+fc1 bias) + mish, then fc2 head ----
    k_bn_mish<<<NUM_ROI, N_GRAPHS, 0, stream>>>(zbuf, fc1b, gamma, beta, zm, N_FC1_PAD);
    k_fc2<<<1, 256, 0, stream>>>(zm, fc2w, fc2b, (float*)d_out);
}
